// CrossModalFusionBlock_31714038513801
// MI455X (gfx1250) — matrix operand order repaired and checked by reading, not compiled, not run on hardware
//
#include <hip/hip_runtime.h>
#include <hip/hip_bf16.h>

typedef __attribute__((ext_vector_type(16))) __bf16  v16bf;
typedef __attribute__((ext_vector_type(8)))  float   v8f;
typedef __attribute__((ext_vector_type(4)))  unsigned int u32x4;
typedef __attribute__((ext_vector_type(4)))  int v4i;

#define NPOS 32768   // B*T
#define HDIM 1024
#define NHEADS 16
#define HEADD 64
#define NMOD 3

#if defined(__HIP_DEVICE_COMPILE__) && defined(__gfx1250__) && \
    __has_builtin(__builtin_amdgcn_global_load_async_to_lds_b128) && \
    __has_builtin(__builtin_amdgcn_s_wait_asynccnt)
#define USE_ASYNC_LDS 1
#else
#define USE_ASYNC_LDS 0
#endif

static __device__ __forceinline__ unsigned short f2bf(float f) {
    unsigned int u = __float_as_uint(f);
    unsigned int r = u + 0x7FFFu + ((u >> 16) & 1u);   // round-to-nearest-even
    return (unsigned short)(r >> 16);
}
static __device__ __forceinline__ float bf2f(unsigned short h) {
    return __uint_as_float(((unsigned int)h) << 16);
}

// ---------------------------------------------------------------------------
// elementwise conversion kernels
// ---------------------------------------------------------------------------
__global__ void cvt_f32_bf16_kernel(const float* __restrict__ in,
                                    unsigned short* __restrict__ out,
                                    long long n) {
    long long i = (long long)blockIdx.x * 256 + threadIdx.x;
    if (i < n) out[i] = f2bf(in[i]);
}

// x2 [N,H] fp32 -> stacked[:,2,:] bf16 (row stride 3*H)
__global__ void cvt_x2_kernel(const float* __restrict__ x2,
                              unsigned short* __restrict__ stacked) {
    long long i = (long long)blockIdx.x * 256 + threadIdx.x;   // over N*H
    long long row = i >> 10;
    int c = (int)(i & 1023);
    stacked[row * (NMOD * HDIM) + 2 * HDIM + c] = f2bf(x2[i]);
}

// qin = bf16(query_token[c] + static_context)
__global__ void qin_kernel(const float* __restrict__ sc,
                           const float* __restrict__ qt,
                           unsigned short* __restrict__ out) {
    long long i = (long long)blockIdx.x * 256 + threadIdx.x;   // over N*H
    out[i] = f2bf(sc[i] + qt[i & 1023]);
}

// ---------------------------------------------------------------------------
// bf16 WMMA GEMM:  C[M,Nout] = A[M,K] * W[Nout,K]^T + bias
// block tile 256x128, 8 waves (4 M x 2 N), wave tile 64x64, K-step 32.
// Double-buffered LDS; tile fill via async global->LDS loads when available.
// ---------------------------------------------------------------------------
#define BM 256
#define BN 128
#define BK 32
#define LDSS 40          // LDS row stride in halfs (32 data + 8 pad -> 80B, 16B aligned)

#define EPI_F32       0
#define EPI_BF16      1
#define EPI_BF16_GELU 2

__global__ __launch_bounds__(256)
void gemm_bf16_kernel(const unsigned short* __restrict__ A,
                      const unsigned short* __restrict__ W,   // [Nout,K] row-major
                      const float* __restrict__ bias,
                      void* __restrict__ out,
                      int K, long long ldout, long long colOff, int mode) {
    __shared__ unsigned short lA[2][BM * LDSS];   // 2 x 20KB
    __shared__ unsigned short lB[2][BN * LDSS];   // 2 x 10KB

    const int tid  = threadIdx.x;
    const int m0   = blockIdx.y * BM;
    const int n0   = blockIdx.x * BN;
    const int wave = tid >> 5;
    const int lane = tid & 31;
    const int wm   = wave & 3;     // wave row (4 along M, 64 rows each)
    const int wn   = wave >> 2;    // wave col (2 along N, 64 cols each)
    const int nlo  = lane & 15;
    const int hi   = lane >> 4;

    // tile fill: rows x 32 halfs; 16B chunks; rows/64 chunks per thread
    auto fillTile = [&](const unsigned short* src, int srcRow0, int k0,
                        unsigned short* dst, int rows) {
        const int per = rows >> 6;             // rows*4 chunks / 256 threads
#pragma unroll
        for (int i = 0; i < 4; ++i) {
            if (i >= per) break;
            int c   = tid + i * 256;
            int row = c >> 2;
            int cc  = c & 3;
            const unsigned short* gp = src + (long long)(srcRow0 + row) * K + k0 + cc * 8;
            unsigned short* lp = dst + row * LDSS + cc * 8;
#if USE_ASYNC_LDS
            v4i* gpn = (v4i*)const_cast<unsigned short*>(gp);
            __builtin_amdgcn_global_load_async_to_lds_b128(
                (__attribute__((address_space(1))) v4i*)gpn,
                (__attribute__((address_space(3))) v4i*)(v4i*)lp, 0, 0);
#else
            *(u32x4*)lp = *(const u32x4*)gp;
#endif
        }
    };
    auto waitFill = [&]() {
#if USE_ASYNC_LDS
        __builtin_amdgcn_s_wait_asynccnt(0);
#endif
    };

    // A frag (16x32): VGPR0-3 <- K = hi*8..hi*8+7 ; VGPR4-7 <- K = 16+hi*8..
    auto loadA = [&](const unsigned short* base, int rowTile) -> v16bf {
        const unsigned short* p = base + (rowTile + nlo) * LDSS;
        union { u32x4 q[2]; v16bf v; } u;
        u.q[0] = *(const u32x4*)(p + hi * 8);
        u.q[1] = *(const u32x4*)(p + 16 + hi * 8);
        return u.v;
    };
    // B frag (32x16): VGPR0-3 <- K = hi*8..hi*8+7 ; VGPR4-7 <- K = 16+hi*8.. (same as A)
    auto loadB = [&](const unsigned short* base, int colTile) -> v16bf {
        const unsigned short* p = base + (colTile + nlo) * LDSS;
        union { u32x4 q[2]; v16bf v; } u;
        u.q[0] = *(const u32x4*)(p + hi * 8);
        u.q[1] = *(const u32x4*)(p + 16 + hi * 8);
        return u.v;
    };

    v8f acc[4][4];
#pragma unroll
    for (int tn = 0; tn < 4; ++tn) {
        float bv = bias ? bias[n0 + wn * 64 + tn * 16 + nlo] : 0.0f;
#pragma unroll
        for (int tm = 0; tm < 4; ++tm)
#pragma unroll
            for (int r = 0; r < 8; ++r) acc[tm][tn][r] = bv;
    }

    const int nk = K / BK;
    fillTile(A, m0, 0, lA[0], BM);
    fillTile(W, n0, 0, lB[0], BN);
    waitFill();
    __syncthreads();

    for (int kt = 0; kt < nk; ++kt) {
        int cur = kt & 1;
        if (kt + 1 < nk) {
            fillTile(A, m0, (kt + 1) * BK, lA[cur ^ 1], BM);
            fillTile(W, n0, (kt + 1) * BK, lB[cur ^ 1], BN);
        }
        v16bf af[4], bf[4];
#pragma unroll
        for (int tm = 0; tm < 4; ++tm) af[tm] = loadA(lA[cur], wm * 64 + tm * 16);
#pragma unroll
        for (int tn = 0; tn < 4; ++tn) bf[tn] = loadB(lB[cur], wn * 64 + tn * 16);
#pragma unroll
        for (int tm = 0; tm < 4; ++tm)
#pragma unroll
            for (int tn = 0; tn < 4; ++tn)
                acc[tm][tn] = __builtin_amdgcn_wmma_f32_16x16x32_bf16(
                    false, af[tm], false, bf[tn], (short)0, acc[tm][tn], false, false);
        if (kt + 1 < nk) waitFill();
        __syncthreads();
    }

    // epilogue: lane holds column nlo, rows hi*8 + r of each 16x16 tile
#pragma unroll
    for (int tm = 0; tm < 4; ++tm) {
#pragma unroll
        for (int tn = 0; tn < 4; ++tn) {
            long long col = n0 + wn * 64 + tn * 16 + nlo;
#pragma unroll
            for (int r = 0; r < 8; ++r) {
                long long row = m0 + wm * 64 + tm * 16 + hi * 8 + r;
                float v = acc[tm][tn][r];
                long long idx = row * ldout + colOff + col;
                if (mode == EPI_F32) {
                    ((float*)out)[idx] = v;
                } else if (mode == EPI_BF16) {
                    ((unsigned short*)out)[idx] = f2bf(v);
                } else {
                    float g = 0.5f * v * (1.0f + erff(v * 0.70710678118654752f));
                    ((unsigned short*)out)[idx] = f2bf(g);
                }
            }
        }
    }
}

// ---------------------------------------------------------------------------
// per-position single-query attention over M=3 modality tokens
// one thread per (position, head); 16 positions x 16 heads per block
// ---------------------------------------------------------------------------
__global__ __launch_bounds__(256)
void attn_kernel(const unsigned short* __restrict__ Q,
                 const unsigned short* __restrict__ K,
                 const unsigned short* __restrict__ V,
                 unsigned short* __restrict__ ctx,
                 float* __restrict__ wout) {
    int tid = threadIdx.x;
    long long n = (long long)blockIdx.x * 16 + (tid >> 4);
    int h = tid & 15;
    const unsigned short* qp = Q + n * HDIM + h * HEADD;

    float s[NMOD];
#pragma unroll
    for (int m = 0; m < NMOD; ++m) {
        const unsigned short* kp = K + (n * NMOD + m) * HDIM + h * HEADD;
        float acc = 0.0f;
#pragma unroll 8
        for (int j = 0; j < HEADD; ++j) acc += bf2f(qp[j]) * bf2f(kp[j]);
        s[m] = acc * 0.125f;   // 1/sqrt(64)
    }
    float mx = fmaxf(s[0], fmaxf(s[1], s[2]));
    float e0 = __expf(s[0] - mx), e1 = __expf(s[1] - mx), e2 = __expf(s[2] - mx);
    float inv = 1.0f / (e0 + e1 + e2);
    float a0 = e0 * inv, a1 = e1 * inv, a2 = e2 * inv;

    // weights = mean over heads (16 threads of one wave half)
    float am[NMOD] = {a0, a1, a2};
#pragma unroll
    for (int m = 0; m < NMOD; ++m) {
        float a = am[m];
        for (int o = 8; o >= 1; o >>= 1) a += __shfl_xor(a, o, 16);
        if (h == 0) wout[n * NMOD + m] = a * (1.0f / 16.0f);
    }

    const unsigned short* v0 = V + (n * NMOD + 0) * HDIM + h * HEADD;
    const unsigned short* v1 = V + (n * NMOD + 1) * HDIM + h * HEADD;
    const unsigned short* v2 = V + (n * NMOD + 2) * HDIM + h * HEADD;
    unsigned short* cp = ctx + n * HDIM + h * HEADD;
#pragma unroll 8
    for (int j = 0; j < HEADD; ++j) {
        float c = a0 * bf2f(v0[j]) + a1 * bf2f(v1[j]) + a2 * bf2f(v2[j]);
        cp[j] = f2bf(c);
    }
}

// ---------------------------------------------------------------------------
// LayerNorm over H=1024, one block (256 threads) per row
// ---------------------------------------------------------------------------
__global__ __launch_bounds__(256)
void ln1_kernel(const float* __restrict__ x, const float* __restrict__ g,
                const float* __restrict__ b, float* __restrict__ y,
                unsigned short* __restrict__ yb) {
    __shared__ float rs[256], rq[256];
    long long row = blockIdx.x;
    const float* xr = x + row * HDIM;
    int tid = threadIdx.x;
    float v[4], s = 0.0f, q = 0.0f;
#pragma unroll
    for (int i = 0; i < 4; ++i) {
        v[i] = xr[tid + i * 256];
        s += v[i]; q += v[i] * v[i];
    }
    rs[tid] = s; rq[tid] = q;
    __syncthreads();
    for (int o = 128; o > 0; o >>= 1) {
        if (tid < o) { rs[tid] += rs[tid + o]; rq[tid] += rq[tid + o]; }
        __syncthreads();
    }
    float mean = rs[0] * (1.0f / HDIM);
    float var  = rq[0] * (1.0f / HDIM) - mean * mean;
    float inv  = rsqrtf(var + 1e-5f);
#pragma unroll
    for (int i = 0; i < 4; ++i) {
        int c = tid + i * 256;
        float o = (v[i] - mean) * inv * g[c] + b[c];
        y[row * HDIM + c]  = o;
        yb[row * HDIM + c] = f2bf(o);
    }
}

// in-place: io = LN(io + ff)
__global__ __launch_bounds__(256)
void ln2_kernel(float* __restrict__ io, const float* __restrict__ ff,
                const float* __restrict__ g, const float* __restrict__ b) {
    __shared__ float rs[256], rq[256];
    long long row = blockIdx.x;
    int tid = threadIdx.x;
    float v[4], s = 0.0f, q = 0.0f;
#pragma unroll
    for (int i = 0; i < 4; ++i) {
        long long idx = row * HDIM + tid + i * 256;
        v[i] = io[idx] + ff[idx];
        s += v[i]; q += v[i] * v[i];
    }
    rs[tid] = s; rq[tid] = q;
    __syncthreads();
    for (int o = 128; o > 0; o >>= 1) {
        if (tid < o) { rs[tid] += rs[tid + o]; rq[tid] += rq[tid + o]; }
        __syncthreads();
    }
    float mean = rs[0] * (1.0f / HDIM);
    float var  = rq[0] * (1.0f / HDIM) - mean * mean;
    float inv  = rsqrtf(var + 1e-5f);
#pragma unroll
    for (int i = 0; i < 4; ++i) {
        int c = tid + i * 256;
        io[row * HDIM + c] = (v[i] - mean) * inv * g[c] + b[c];
    }
}

// ---------------------------------------------------------------------------
// host
// ---------------------------------------------------------------------------
extern "C" void kernel_launch(void* const* d_in, const int* in_sizes, int n_in,
                              void* d_out, int out_size, void* d_ws, size_t ws_size,
                              hipStream_t stream) {
    const float* x0  = (const float*)d_in[0];
    const float* x1  = (const float*)d_in[1];
    const float* x2  = (const float*)d_in[2];
    const float* sc  = (const float*)d_in[3];
    const float* Wp0 = (const float*)d_in[4];
    const float* bp0 = (const float*)d_in[5];
    const float* Wp1 = (const float*)d_in[6];
    const float* bp1 = (const float*)d_in[7];
    const float* Wq  = (const float*)d_in[8];
    const float* bq  = (const float*)d_in[9];
    const float* Wk  = (const float*)d_in[10];
    const float* bk  = (const float*)d_in[11];
    const float* Wv  = (const float*)d_in[12];
    const float* bv  = (const float*)d_in[13];
    const float* Wo  = (const float*)d_in[14];
    const float* bo  = (const float*)d_in[15];
    const float* qt  = (const float*)d_in[16];
    const float* g1  = (const float*)d_in[17];
    const float* be1 = (const float*)d_in[18];
    const float* g2  = (const float*)d_in[19];
    const float* be2 = (const float*)d_in[20];
    const float* W1  = (const float*)d_in[21];
    const float* b1  = (const float*)d_in[22];
    const float* W2  = (const float*)d_in[23];
    const float* b2  = (const float*)d_in[24];

    char* ws = (char*)d_ws;
    size_t off = 0;
    auto alloc = [&](size_t bytes) { size_t o = off; off += (bytes + 255) & ~(size_t)255; return o; };

    const size_t NH_ = (size_t)NPOS * HDIM;
    unsigned short* Wb       = (unsigned short*)(ws + alloc(13369344ull * 2));
    unsigned short* x0b      = (unsigned short*)(ws + alloc((size_t)NPOS * 256 * 2));
    unsigned short* x1b      = (unsigned short*)(ws + alloc((size_t)NPOS * 512 * 2));
    unsigned short* qinb     = (unsigned short*)(ws + alloc(NH_ * 2));
    unsigned short* Qb       = (unsigned short*)(ws + alloc(NH_ * 2));
    unsigned short* stackedb = (unsigned short*)(ws + alloc(NH_ * NMOD * 2));
    unsigned short* Kb       = (unsigned short*)(ws + alloc(NH_ * NMOD * 2));
    unsigned short* Vb       = (unsigned short*)(ws + alloc(NH_ * NMOD * 2));
    unsigned short* ctxb     = (unsigned short*)(ws + alloc(NH_ * 2));
    unsigned short* fusedb   = (unsigned short*)(ws + alloc(NH_ * 2));
    (void)ws_size; (void)in_sizes; (void)n_in; (void)out_size;

    // weight sub-buffers (bf16)
    unsigned short* Wp0b = Wb;
    unsigned short* Wp1b = Wb + 262144;
    unsigned short* Wqb  = Wb + 786432;
    unsigned short* Wkb  = Wb + 1835008;
    unsigned short* Wvb  = Wb + 2883584;
    unsigned short* Wob  = Wb + 3932160;
    unsigned short* W1b  = Wb + 4980736;
    unsigned short* W2b  = Wb + 9175040;

    // aliases of dead buffers
    float*          attn_out = (float*)stackedb;     // 128MB in 192MB
    unsigned short* h1b      = Kb;                   // 256MB spanning Kb+Vb (384MB)
    float*          ffbuf    = (float*)qinb;         // 128MB spanning qinb+Qb
    float*          fused    = (float*)d_out;        // [N,H] fp32, finalized by ln2
    float*          wout     = (float*)d_out + NH_;  // [N,3] fp32

    dim3 blk(256);
    auto cvt = [&](const float* in, unsigned short* out, long long n) {
        cvt_f32_bf16_kernel<<<dim3((unsigned)((n + 255) / 256)), blk, 0, stream>>>(in, out, n);
    };

    // 1. convert weights + activations to bf16
    cvt(Wp0, Wp0b, 262144);   cvt(Wp1, Wp1b, 524288);
    cvt(Wq, Wqb, 1048576);    cvt(Wk, Wkb, 1048576);
    cvt(Wv, Wvb, 1048576);    cvt(Wo, Wob, 1048576);
    cvt(W1, W1b, 4194304);    cvt(W2, W2b, 4194304);
    cvt(x0, x0b, (long long)NPOS * 256);
    cvt(x1, x1b, (long long)NPOS * 512);
    cvt_x2_kernel<<<dim3((unsigned)(NH_ / 256)), blk, 0, stream>>>(x2, stackedb);
    qin_kernel<<<dim3((unsigned)(NH_ / 256)), blk, 0, stream>>>(sc, qt, qinb);

    auto gemm = [&](const unsigned short* A, const unsigned short* W, const float* bias,
                    void* out, int M, int Nout, int K, long long ldout, long long colOff,
                    int mode) {
        dim3 grid(Nout / BN, M / BM);
        gemm_bf16_kernel<<<grid, blk, 0, stream>>>(A, W, bias, out, K, ldout, colOff, mode);
    };

    // 2. modality projections into stacked[:, {0,1}, :]
    gemm(x0b, Wp0b, bp0, stackedb, NPOS, HDIM, 256, NMOD * HDIM, 0,     EPI_BF16);
    gemm(x1b, Wp1b, bp1, stackedb, NPOS, HDIM, 512, NMOD * HDIM, HDIM, EPI_BF16);

    // 3. Q / K / V projections
    gemm(qinb,     Wqb, bq, Qb, NPOS,        HDIM, HDIM, HDIM, 0, EPI_BF16);
    gemm(stackedb, Wkb, bk, Kb, NPOS * NMOD, HDIM, HDIM, HDIM, 0, EPI_BF16);
    gemm(stackedb, Wvb, bv, Vb, NPOS * NMOD, HDIM, HDIM, HDIM, 0, EPI_BF16);

    // 4. tiny per-position softmax attention over 3 modality tokens
    attn_kernel<<<dim3(NPOS / 16), blk, 0, stream>>>(Qb, Kb, Vb, ctxb, wout);

    // 5. output projection (fp32 out), LN1 (keeps fp32 in d_out + bf16 copy)
    gemm(ctxb, Wob, bo, attn_out, NPOS, HDIM, HDIM, HDIM, 0, EPI_F32);
    ln1_kernel<<<dim3(NPOS), blk, 0, stream>>>(attn_out, g1, be1, fused, fusedb);

    // 6. FFN: gelu(fused @ W1^T + b1) @ W2^T + b2
    gemm(fusedb, W1b, b1, h1b,   NPOS, 4 * HDIM, HDIM,     4 * HDIM, 0, EPI_BF16_GELU);
    gemm(h1b,    W2b, b2, ffbuf, NPOS, HDIM,     4 * HDIM, HDIM,     0, EPI_F32);

    // 7. final residual LayerNorm, in place in d_out
    ln2_kernel<<<dim3(NPOS), blk, 0, stream>>>(fused, ffbuf, g2, be2);
}
